// Label_Attention_32315333935653
// MI455X (gfx1250) — matrix operand order repaired and checked by reading, not compiled, not run on hardware
//
#include <hip/hip_runtime.h>
#include <cstdint>

// ---------------------------------------------------------------------------
// Label attention: out = softmax(X @ W^T) @ W
//   X: [16384, 1024] f32, W: [1024, 1024] f32, out: [16384, 1024] f32
// bf16 WMMA path (v_wmma_f32_16x16x32_bf16).
//   - A panels (32 rows) staged in LDS, shared by all 8 waves of a block
//   - kernel2 A panel staged by the Tensor Data Mover (tensor_load_to_lds)
//   - each wave computes 2 M-tiles x 8 N-tiles -> B frag reused 2x
//   - softmax fused into GEMM1, block-local over L = 1024
// ---------------------------------------------------------------------------

#define MTOT 16384   // B*S
#define DK   1024    // D (= K of GEMM1, N of GEMM2)
#define LN   1024    // L (= N of GEMM1, K of GEMM2)

typedef __attribute__((ext_vector_type(16))) __bf16 v16bf;
typedef __attribute__((ext_vector_type(8)))  __bf16 v8bf;
typedef __attribute__((ext_vector_type(4)))  __bf16 v4bf;
typedef __attribute__((ext_vector_type(8)))  float  v8f;

typedef __attribute__((ext_vector_type(4))) unsigned int u32x4;
typedef __attribute__((ext_vector_type(8))) int          i32x8;
typedef __attribute__((ext_vector_type(4))) int          i32x4;

// ---------------------------------------------------------------------------
// Kernel 0: W f32 -> Wb bf16 (row-major [L][D]) and Wt bf16 (transposed [D][L])
// ---------------------------------------------------------------------------
__global__ void __launch_bounds__(256) k_convert_w(const float* __restrict__ W,
                                                   __bf16* __restrict__ Wb,
                                                   __bf16* __restrict__ Wt) {
    const size_t i = (size_t)blockIdx.x * blockDim.x + threadIdx.x;
    if (i < (size_t)LN * DK) {
        const int l = (int)(i >> 10);      // row of W
        const int d = (int)(i & (DK - 1)); // col of W
        const __bf16 v = (__bf16)W[i];
        Wb[i] = v;
        Wt[(size_t)d * LN + l] = v;        // Wt[d][l] = W[l][d]
    }
}

// ---------------------------------------------------------------------------
// Shared GEMM mainloop: 2 M-tiles (32 rows in LDS) x 8 N-tiles per wave,
// K = 1024 in steps of 32.
//   As   : LDS bf16 [32][1024] row-major A panel
//   Brows: global bf16 base of this wave's 128 B-columns (each 1024-long K row)
//   acc[mt][t]: f32 C tile, rows [16mt,16mt+16) x cols [16t,16t+16)
// A frag (16x32 bf16): lanes 0-15 hold K {0..15},
//   lanes 16-31 hold K {16..31} -> two 16B ds loads per lane.
// B frag (32x16 bf16): lane holds col = lane&15, K run = (lane>>4)*16 .. +15
//   -> one contiguous 32B global load per lane.
// ---------------------------------------------------------------------------
__device__ __forceinline__ void gemm_k1024(const __bf16* As,
                                           const __bf16* __restrict__ Brows,
                                           v8f acc[2][8], int lane) {
    const int col  = lane & 15;
    const int half = lane >> 4;
    const __bf16* a0 = As + (size_t)col * DK + half * 16;        // rows 0..15
    const __bf16* a1 = As + (size_t)(16 + col) * DK + half * 16; // rows 16..31

#pragma unroll 2
    for (int kc = 0; kc < 1024; kc += 32) {
        // prefetch B stream a few chunks ahead (speculative; OOB dropped)
        __builtin_prefetch(Brows + (size_t)col * DK + kc + 512, 0, 1);

        v16bf af[2];
        {
            const v8bf lo0 = *(const v8bf*)(a0 + kc);
            const v8bf hi0 = *(const v8bf*)(a0 + kc + 8);
            const v8bf lo1 = *(const v8bf*)(a1 + kc);
            const v8bf hi1 = *(const v8bf*)(a1 + kc + 8);
#pragma unroll
            for (int e = 0; e < 8; ++e) {
                af[0][e] = lo0[e]; af[0][e + 8] = hi0[e];
                af[1][e] = lo1[e]; af[1][e + 8] = hi1[e];
            }
        }
#pragma unroll
        for (int t = 0; t < 8; ++t) {
            const v16bf b =
                *(const v16bf*)(Brows + (size_t)(t * 16 + col) * DK + kc + half * 16);
            acc[0][t] = __builtin_amdgcn_wmma_f32_16x16x32_bf16(
                false, af[0], false, b, (short)0, acc[0][t], false, false);
            acc[1][t] = __builtin_amdgcn_wmma_f32_16x16x32_bf16(
                false, af[1], false, b, (short)0, acc[1][t], false, false);
        }
    }
}

// ---------------------------------------------------------------------------
// Kernel 1: scores = X @ Wb^T (f32->bf16 conversion fused into LDS staging),
// fused row softmax, emit P (bf16).
// Block = 256 threads = 8 waves; block owns 32 rows; wave w owns labels
// [128w, 128w+128). Softmax reduced across waves through LDS.
// C-tile layout (32-bit 16x16): element (m = j + 8*half, n = lane&15) in VGPR j.
// ---------------------------------------------------------------------------
__global__ void __launch_bounds__(256) k_scores_softmax(const float* __restrict__ X,
                                                        const __bf16* __restrict__ Wb,
                                                        __bf16* __restrict__ P) {
    const int tid  = threadIdx.x;
    const int lane = tid & 31;
    const int wave = tid >> 5;
    const int col  = lane & 15;
    const int half = lane >> 4;
    const int row0 = blockIdx.x * 32;

    __shared__ __bf16 As[32 * DK];    // 64 KB staged A panel
    __shared__ float  red[2][8][32];  // [max|sum][wave][row-in-block]

    // ---- stage + convert A panel: 32 rows of X (f32) -> LDS bf16 -----------
    for (int i = tid * 4; i < 32 * DK; i += 256 * 4) {
        const float4 f = *(const float4*)(X + (size_t)row0 * DK + i);
        v4bf o;
        o[0] = (__bf16)f.x; o[1] = (__bf16)f.y;
        o[2] = (__bf16)f.z; o[3] = (__bf16)f.w;
        *(v4bf*)(&As[i]) = o;
    }
    __syncthreads();

    v8f acc[2][8] = {};
    gemm_k1024(As, Wb + (size_t)wave * 128 * DK, acc, lane);

    // ---- per-row max over this wave's 128 cols -----------------------------
    float rowm[2][8];
#pragma unroll
    for (int mt = 0; mt < 2; ++mt) {
#pragma unroll
        for (int j = 0; j < 8; ++j) {
            float v = acc[mt][0][j];
#pragma unroll
            for (int t = 1; t < 8; ++t) v = fmaxf(v, acc[mt][t][j]);
#pragma unroll
            for (int off = 8; off >= 1; off >>= 1)
                v = fmaxf(v, __shfl_xor(v, off, 32));  // stays in 16-lane half-group
            rowm[mt][j] = v;
        }
    }
    if (col == 0) {
#pragma unroll
        for (int mt = 0; mt < 2; ++mt)
#pragma unroll
            for (int j = 0; j < 8; ++j)
                red[0][wave][mt * 16 + j + 8 * half] = rowm[mt][j];
    }
    __syncthreads();
#pragma unroll
    for (int mt = 0; mt < 2; ++mt) {
#pragma unroll
        for (int j = 0; j < 8; ++j) {
            const int m = mt * 16 + j + 8 * half;
            float v = red[0][0][m];
#pragma unroll
            for (int w2 = 1; w2 < 8; ++w2) v = fmaxf(v, red[0][w2][m]);
            rowm[mt][j] = v;
        }
    }

    // ---- exp + per-row sum -------------------------------------------------
    float rsum[2][8];
#pragma unroll
    for (int mt = 0; mt < 2; ++mt) {
#pragma unroll
        for (int j = 0; j < 8; ++j) {
            float s = 0.0f;
#pragma unroll
            for (int t = 0; t < 8; ++t) {
                const float e = __expf(acc[mt][t][j] - rowm[mt][j]);
                acc[mt][t][j] = e;
                s += e;
            }
#pragma unroll
            for (int off = 8; off >= 1; off >>= 1)
                s += __shfl_xor(s, off, 32);
            rsum[mt][j] = s;
        }
    }
    if (col == 0) {
#pragma unroll
        for (int mt = 0; mt < 2; ++mt)
#pragma unroll
            for (int j = 0; j < 8; ++j)
                red[1][wave][mt * 16 + j + 8 * half] = rsum[mt][j];
    }
    __syncthreads();
#pragma unroll
    for (int mt = 0; mt < 2; ++mt) {
#pragma unroll
        for (int j = 0; j < 8; ++j) {
            const int m = mt * 16 + j + 8 * half;
            float v = 0.0f;
#pragma unroll
            for (int w2 = 0; w2 < 8; ++w2) v += red[1][w2][m];
            rowm[mt][j] = 1.0f / v;   // reuse as inverse denominator
        }
    }

    // ---- normalize + store P as bf16 (lanes 0-15 -> 32B coalesced runs) -----
#pragma unroll
    for (int mt = 0; mt < 2; ++mt) {
#pragma unroll
        for (int t = 0; t < 8; ++t) {
#pragma unroll
            for (int j = 0; j < 8; ++j) {
                const size_t r = (size_t)(row0 + mt * 16 + j + 8 * half);
                P[r * LN + wave * 128 + t * 16 + col] =
                    (__bf16)(acc[mt][t][j] * rowm[mt][j]);
            }
        }
    }
}

// ---------------------------------------------------------------------------
// Kernel 2: out = P @ W  (A = P bf16 staged in LDS via the Tensor Data Mover,
// B = Wt bf16 [D][L]), f32 out.
// Same tiling: block owns 32 rows, wave w owns output cols [128w, 128w+128).
// The 32x1024 bf16 P panel is one contiguous 64 KB region -> a 1-D TDM tile
// (data_size = 8B, tile_dim0 = tensor_dim0 = 8192). Wave 0 issues the DMA,
// waits TENSORcnt==0, and the workgroup barrier publishes the panel.
// ---------------------------------------------------------------------------
__global__ void __launch_bounds__(256) k_out_gemm(const __bf16* __restrict__ P,
                                                  const __bf16* __restrict__ Wt,
                                                  float* __restrict__ OUT) {
    const int tid  = threadIdx.x;
    const int lane = tid & 31;
    const int wave = tid >> 5;
    const int col  = lane & 15;
    const int half = lane >> 4;
    const int row0 = blockIdx.x * 32;

    __shared__ __bf16 As[32 * LN];    // 64 KB staged P panel

    if (tid < 32) {
        // ---- Tensor DMA descriptor (ISA 8.3 / 8.4) -------------------------
        const unsigned long long gaddr =
            (unsigned long long)(uintptr_t)(P + (size_t)row0 * LN);
        const unsigned int lds = (unsigned int)(uintptr_t)(&As[0]);

        u32x4 g0 = {};
        g0[0] = 1u;                                    // count=1, user descriptor
        g0[1] = lds;                                   // lds_addr[31:0]
        g0[2] = (unsigned int)gaddr;                   // global_addr[31:0]
        g0[3] = (unsigned int)((gaddr >> 32) & 0x01FFFFFFu) |
                0x80000000u;                           // addr[56:32] | type=2

        i32x8 g1 = {};
        g1[0] = (int)(3u << 16);                       // data_size = 8 bytes
        g1[1] = (int)(8192u << 16);                    // tensor_dim0[15:0]
        g1[2] = 0;                                     // tensor_dim0[31:16]=0, dim1=0
        g1[3] = (int)(8192u << 16);                    // tile_dim0 = 8192
        g1[4] = 0;                                     // tile_dim1/2 = 0 (1-D)
        g1[5] = 8192;                                  // tensor_dim0_stride[31:0]
        g1[6] = 0;
        g1[7] = 0;

        i32x4 g2 = {};
        i32x4 g3 = {};
#if defined(__clang_major__) && (__clang_major__ >= 23)
        i32x8 g4 = {};
        __builtin_amdgcn_tensor_load_to_lds(g0, g1, g2, g3, g4, 0);
#else
        __builtin_amdgcn_tensor_load_to_lds(g0, g1, g2, g3, 0);
#endif
        __builtin_amdgcn_s_wait_tensorcnt(0);
    }
    __syncthreads();

    v8f acc[2][8] = {};
    gemm_k1024(As, Wt + (size_t)wave * 128 * LN, acc, lane);

#pragma unroll
    for (int mt = 0; mt < 2; ++mt) {
#pragma unroll
        for (int t = 0; t < 8; ++t) {
#pragma unroll
            for (int j = 0; j < 8; ++j) {
                const size_t r = (size_t)(row0 + mt * 16 + j + 8 * half);
                OUT[r * DK + wave * 128 + t * 16 + col] = acc[mt][t][j];
            }
        }
    }
}

// ---------------------------------------------------------------------------
// Workspace layout (bytes):
//   [0,    2MB)  Wb  bf16 [1024][1024]
//   [2MB,  4MB)  Wt  bf16 [1024][1024] (transposed)
//   [4MB, 36MB)  P   bf16 [16384][1024]
// ---------------------------------------------------------------------------
extern "C" void kernel_launch(void* const* d_in, const int* in_sizes, int n_in,
                              void* d_out, int out_size, void* d_ws, size_t ws_size,
                              hipStream_t stream) {
    (void)in_sizes; (void)n_in; (void)out_size; (void)ws_size;

    const float* X = (const float*)d_in[0];
    const float* W = (const float*)d_in[1];
    float* OUT     = (float*)d_out;

    char* ws = (char*)d_ws;
    __bf16* Wb = (__bf16*)(ws);
    __bf16* Wt = (__bf16*)(ws + (size_t)2 * 1024 * 1024);
    __bf16* Pm = (__bf16*)(ws + (size_t)4 * 1024 * 1024);

    // 0) weight precision staging (+ transpose for GEMM2 B fragments)
    k_convert_w<<<((size_t)LN * DK) / 256, 256, 0, stream>>>(W, Wb, Wt);

    // 1) logits + fused softmax -> P  (X f32->bf16 fused into LDS staging)
    k_scores_softmax<<<MTOT / 32, 256, 0, stream>>>(X, Wb, Pm);

    // 2) out = P @ W  (P panel staged by TDM)
    k_out_gemm<<<MTOT / 32, 256, 0, stream>>>(Pm, Wt, OUT);
}
